// SymNetDP_27101243638023
// MI455X (gfx1250) — compile-verified, matrix operand order checked by reading, not run on hardware
//
#include <hip/hip_runtime.h>
#include <hip/hip_bf16.h>

// ---------------------------------------------------------------------------
// SymNet group-conv pipeline for MI455X (gfx1250, wave32, WMMA).
// B=64, S=4096, NGB=13, NG=48, DIM=3, NCH=[8,8,1], NSHELLS=8
// ---------------------------------------------------------------------------

#define B_   64
#define S_   4096
#define NGB_ 13
#define NG_  48

typedef __attribute__((ext_vector_type(16))) __bf16 v16bf;
typedef __attribute__((ext_vector_type(8)))  float  v8f;

union FragU {        // 16 bf16 = 32 bytes = two 16-byte vector loads
    v16bf v;
    uint4 u[2];
};

__device__ __forceinline__ float softplus_f(float x) {
    // softplus = max(x,0) + ln2 * log2(1 + 2^(-|x|*log2e))
    // raw v_exp_f32 / v_log_f32, no denormal guards (arg of log2 is in [1,2])
    float t = __builtin_amdgcn_exp2f(fabsf(x) * -1.44269504088896340736f);
    return fmaxf(x, 0.0f) + 0.69314718055994530942f * __builtin_amdgcn_logf(1.0f + t);
}

// ---------------------------------------------------------------------------
// Prep: build group-rotated weight matrices GW (bf16, row-major, K padded)
// and the fused vector-channel matrix wsum[3][13] = (1/48)Σ_g gdiags·P.
// ---------------------------------------------------------------------------
__global__ void prep_kernel(const float* __restrict__ Psi0,
                            const float* __restrict__ Psi1,
                            const float* __restrict__ Psi2,
                            const float* __restrict__ wtVC,
                            const float* __restrict__ gdiags,
                            const int*   __restrict__ GnnPerms,
                            __bf16* __restrict__ GW0,   // 384 x 32
                            __bf16* __restrict__ GW1,   // 384 x 128
                            __bf16* __restrict__ GW2,   //  48 x 128
                            float*  __restrict__ wsum)  // 3 x 13
{
    const int tid = threadIdx.x;

    // GW0[o*48+g][k] = Psi0[o][0][perm[g][k]]  (K=13 padded to 32)
    for (int idx = tid; idx < 384 * 32; idx += 256) {
        int m = idx >> 5, k = idx & 31;
        int o = m / NG_, g = m % NG_;
        float v = 0.0f;
        if (k < NGB_) v = Psi0[o * NGB_ + GnnPerms[g * NGB_ + k]];
        GW0[idx] = (__bf16)v;
    }
    // GW1[o*48+g][c*13+j] = Psi1[o][c][perm[g][j]]  (K=104 padded to 128)
    for (int idx = tid; idx < 384 * 128; idx += 256) {
        int m = idx >> 7, k = idx & 127;
        int o = m / NG_, g = m % NG_;
        float v = 0.0f;
        if (k < 8 * NGB_) {
            int c = k / NGB_, j = k - c * NGB_;
            v = Psi1[(o * 8 + c) * NGB_ + GnnPerms[g * NGB_ + j]];
        }
        GW1[idx] = (__bf16)v;
    }
    // GW2[g][c*13+j] = Psi2[0][c][perm[g][j]]
    for (int idx = tid; idx < 48 * 128; idx += 256) {
        int g = idx >> 7, k = idx & 127;
        float v = 0.0f;
        if (k < 8 * NGB_) {
            int c = k / NGB_, j = k - c * NGB_;
            v = Psi2[c * NGB_ + GnnPerms[g * NGB_ + j]];
        }
        GW2[idx] = (__bf16)v;
    }
    // wsum[d][n] = (1/48) Σ_g Σ_d' gdiags[g*3+d][g*3+d'] * wtVC[d'][perm[g][n]]
    for (int idx = tid; idx < 3 * NGB_; idx += 256) {
        int d = idx / NGB_, n = idx - d * NGB_;
        float acc = 0.0f;
        for (int g = 0; g < NG_; ++g) {
            int p = GnnPerms[g * NGB_ + n];
            for (int dp = 0; dp < 3; ++dp)
                acc += gdiags[(g * 3 + d) * 144 + (g * 3 + dp)] * wtVC[dp * NGB_ + p];
        }
        wsum[idx] = acc * (1.0f / (float)NG_);
    }
}

// ---------------------------------------------------------------------------
// Fused rearrange + group-conv + softplus + orbit-mean.
//   xin  : (B, NCIN, S) f32       gw : (NCOUT*48, KP) bf16 row-major
//   xout : (B, NCOUT, S) f32
// Block = 256 threads (8 wave32). Workgroup handles one b and 128 sites.
// GW staged once per WG into padded LDS. Each wave owns a 16-site column
// tile; B-fragments are hoisted into registers once per wave, and each
// output channel o runs THREE independent v_wmma_f32_16x16x32_bf16 chains.
// ---------------------------------------------------------------------------
template <int NCIN, int NCOUT, int KP>
__global__ void gconv_kernel(const float* __restrict__ xin,
                             const int*   __restrict__ nnsites,
                             const __bf16* __restrict__ gw,
                             const float* __restrict__ bias,
                             float* __restrict__ xout)
{
    constexpr int K       = NCIN * NGB_;        // 13 or 104
    constexpr int M       = NCOUT * NG_;        // 384 or 48
    constexpr int XSTRIDE = KP + 8;             // kill 16-way LDS bank conflicts
    constexpr int KSHIFT  = (KP == 32) ? 5 : 7;
    constexpr int NKC     = KP / 32;            // k-chunks (1 or 4)

    __shared__ __bf16 XT[128 * XSTRIDE];        // [site][k] activations, bf16
    __shared__ __bf16 WT[M * XSTRIDE];          // staged GW, padded stride

    const int tid = threadIdx.x;
    const int b   = blockIdx.y;
    const int s0  = blockIdx.x * 128;

    // warm L2/WGP$ for the weight staging stream (global_prefetch_b8)
    __builtin_prefetch(gw + (size_t)tid * ((M * KP) / 256), 0, 0);

    // ---- stage GW (row-major, 16B chunks) into padded LDS
    for (int e = tid; e < (M * KP) / 8; e += 256) {
        int e8 = e * 8;
        int m = e8 >> KSHIFT, k = e8 & (KP - 1);
        *(uint4*)(&WT[m * XSTRIDE + k]) = *(const uint4*)(gw + (size_t)m * KP + k);
    }

    // ---- gather (rearrange) the activation tile into LDS, convert to bf16
    for (int e = tid; e < 128 * KP; e += 256) {
        int sl = e >> KSHIFT, kk = e & (KP - 1);
        float v = 0.0f;
        if (kk < K) {
            int c = kk / NGB_, j = kk - c * NGB_;
            int site = nnsites[j * S_ + s0 + sl];
            v = xin[((size_t)b * NCIN + c) * S_ + site];
        }
        XT[sl * XSTRIDE + kk] = (__bf16)v;
    }
    __syncthreads();

    const int lane = tid & 31;
    const int w    = tid >> 5;      // wave id -> 16-site column tile
    const int n    = lane & 15;
    const int hi   = lane >> 4;
    const int xoff = (w * 16 + n) * XSTRIDE;    // B-fragment row base in XT

    // ---- hoist all B-fragments for this wave's 16 columns into registers
    FragU bx[NKC];
    #pragma unroll
    for (int kc = 0; kc < NKC; ++kc) {
        const int ka = kc * 32 + hi * 8;        // ISA 16-bit A/B lane K pattern
        bx[kc].u[0] = *(const uint4*)(&XT[xoff + ka]);
        bx[kc].u[1] = *(const uint4*)(&XT[xoff + ka + 16]);
    }

    for (int o = 0; o < NCOUT; ++o) {
        v8f acc0 = {}, acc1 = {}, acc2 = {};    // 3 independent WMMA chains
        const int m0 = (o * NG_ + n) * XSTRIDE; // A rows: o*48 + {0,16,32} + n
        #pragma unroll
        for (int kc = 0; kc < NKC; ++kc) {
            const int ka = kc * 32 + hi * 8;
            FragU a0, a1, a2;
            a0.u[0] = *(const uint4*)(&WT[m0 + ka]);
            a0.u[1] = *(const uint4*)(&WT[m0 + ka + 16]);
            a1.u[0] = *(const uint4*)(&WT[m0 + 16 * XSTRIDE + ka]);
            a1.u[1] = *(const uint4*)(&WT[m0 + 16 * XSTRIDE + ka + 16]);
            a2.u[0] = *(const uint4*)(&WT[m0 + 32 * XSTRIDE + ka]);
            a2.u[1] = *(const uint4*)(&WT[m0 + 32 * XSTRIDE + ka + 16]);
            acc0 = __builtin_amdgcn_wmma_f32_16x16x32_bf16(
                false, a0.v, false, bx[kc].v, (short)0, acc0, false, false);
            acc1 = __builtin_amdgcn_wmma_f32_16x16x32_bf16(
                false, a1.v, false, bx[kc].v, (short)0, acc1, false, false);
            acc2 = __builtin_amdgcn_wmma_f32_16x16x32_bf16(
                false, a2.v, false, bx[kc].v, (short)0, acc2, false, false);
        }
        const float bo = bias[o];               // loop-uniform -> s_load
        float part = 0.0f;
        #pragma unroll
        for (int r = 0; r < 8; ++r)             // lane holds rows m = r + 8*hi
            part += softplus_f(acc0[r] + bo) +
                    softplus_f(acc1[r] + bo) +
                    softplus_f(acc2[r] + bo);
        part += __shfl_xor(part, 16);           // combine m<8 with m>=8 halves
        if (hi == 0)
            xout[((size_t)b * NCOUT + o) * S_ + s0 + w * 16 + n] =
                part * (1.0f / (float)NG_);
    }
}

// ---------------------------------------------------------------------------
// Vector channel: out[b][d] = (1/S) Σ_s SW[shell[s]] Σ_j wsum[d][j]·x2[b][NN[j,s]]
// ---------------------------------------------------------------------------
__global__ void vc_reduce_kernel(const float* __restrict__ out2,
                                 const float* __restrict__ wsum,
                                 const float* __restrict__ shellw,
                                 const int*   __restrict__ s2sh,
                                 const int*   __restrict__ nnsites,
                                 float* __restrict__ out)
{
    __shared__ float red[3][256];
    __shared__ float ws[3 * NGB_];
    const int b = blockIdx.x, tid = threadIdx.x;
    if (tid < 3 * NGB_) ws[tid] = wsum[tid];
    __syncthreads();

    float a0 = 0.f, a1 = 0.f, a2 = 0.f;
    for (int s = tid; s < S_; s += 256) {
        float sh = shellw[s2sh[s]];
        float t0 = 0.f, t1 = 0.f, t2 = 0.f;
        #pragma unroll
        for (int j = 0; j < NGB_; ++j) {
            float g = out2[(size_t)b * S_ + nnsites[j * S_ + s]];
            t0 += ws[j] * g;
            t1 += ws[NGB_ + j] * g;
            t2 += ws[2 * NGB_ + j] * g;
        }
        a0 += sh * t0; a1 += sh * t1; a2 += sh * t2;
    }
    red[0][tid] = a0; red[1][tid] = a1; red[2][tid] = a2;
    __syncthreads();
    for (int st = 128; st > 0; st >>= 1) {
        if (tid < st) {
            red[0][tid] += red[0][tid + st];
            red[1][tid] += red[1][tid + st];
            red[2][tid] += red[2][tid + st];
        }
        __syncthreads();
    }
    if (tid == 0) {
        out[b * 3 + 0] = red[0][0] * (1.0f / (float)S_);
        out[b * 3 + 1] = red[1][0] * (1.0f / (float)S_);
        out[b * 3 + 2] = red[2][0] * (1.0f / (float)S_);
    }
}

// ---------------------------------------------------------------------------
extern "C" void kernel_launch(void* const* d_in, const int* in_sizes, int n_in,
                              void* d_out, int out_size, void* d_ws, size_t ws_size,
                              hipStream_t stream)
{
    (void)in_sizes; (void)n_in; (void)out_size; (void)ws_size;

    const float* InStates = (const float*)d_in[0];
    const float* Psi0     = (const float*)d_in[1];
    const float* bias0    = (const float*)d_in[2];
    const float* Psi1     = (const float*)d_in[3];
    const float* bias1    = (const float*)d_in[4];
    const float* Psi2     = (const float*)d_in[5];
    const float* bias2    = (const float*)d_in[6];
    const float* wtVC     = (const float*)d_in[7];
    const float* ShellW   = (const float*)d_in[8];
    const float* gdiags   = (const float*)d_in[9];
    const int*   GnnPerms = (const int*)d_in[10];
    const int*   NNSites  = (const int*)d_in[11];
    const int*   S2Sh     = (const int*)d_in[12];

    char*  ws  = (char*)d_ws;
    size_t off = 0;
    auto alloc = [&](size_t bytes) -> void* {
        void* p = ws + off;
        off = (off + bytes + 255) & ~(size_t)255;
        return p;
    };
    __bf16* GW0  = (__bf16*)alloc((size_t)384 * 32 * 2);
    __bf16* GW1  = (__bf16*)alloc((size_t)384 * 128 * 2);
    __bf16* GW2  = (__bf16*)alloc((size_t)48 * 128 * 2);
    float*  wsum = (float*) alloc((size_t)3 * NGB_ * 4);
    float*  out0 = (float*) alloc((size_t)B_ * 8 * S_ * 4);   // 8 MB (L2-resident)
    float*  out1 = (float*) alloc((size_t)B_ * 8 * S_ * 4);   // 8 MB
    float*  out2 = (float*) alloc((size_t)B_ * 1 * S_ * 4);   // 1 MB

    prep_kernel<<<1, 256, 0, stream>>>(Psi0, Psi1, Psi2, wtVC, gdiags, GnnPerms,
                                       GW0, GW1, GW2, wsum);

    dim3 grid(S_ / 128, B_);
    gconv_kernel<1, 8, 32 ><<<grid, 256, 0, stream>>>(InStates, NNSites, GW0, bias0, out0);
    gconv_kernel<8, 8, 128><<<grid, 256, 0, stream>>>(out0,     NNSites, GW1, bias1, out1);
    gconv_kernel<8, 1, 128><<<grid, 256, 0, stream>>>(out1,     NNSites, GW2, bias2, out2);

    vc_reduce_kernel<<<B_, 256, 0, stream>>>(out2, wsum, ShellW, S2Sh, NNSites,
                                             (float*)d_out);
    (void)bias2;
}
